// MAB_22110491640673
// MI455X (gfx1250) — matrix operand order repaired and checked by reading, not compiled, not run on hardware
//
#include <hip/hip_runtime.h>

// ---------------------------------------------------------------------------
// MAB (multihead attention block) for MI455X / gfx1250, wave32 + WMMA bf16.
// Nq = Nk = 4096, D = 1024, H = 16, dh = 64, scale = 1/32.
// ---------------------------------------------------------------------------

typedef __attribute__((ext_vector_type(16))) __bf16 v16bf;
typedef __attribute__((ext_vector_type(8)))  float  v8f;

struct __align__(16) U4 { unsigned int x, y, z, w; };
struct __align__(16) F4 { float x, y, z, w; };

union BFrag {            // 16 bf16 = 8 VGPRs (A or B operand of wmma 16x16x32)
  v16bf v;
  U4 q[2];
};
union FFrag {            // 16x16 f32 C/D fragment = 8 VGPRs
  v8f v;
  float f[8];
};

__device__ __forceinline__ unsigned short f2bf(float f) {
  unsigned int u = __float_as_uint(f);
  u += 0x7FFFu + ((u >> 16) & 1u);          // round-to-nearest-even
  return (unsigned short)(u >> 16);
}
__device__ __forceinline__ unsigned int pack_bf2(float lo, float hi) {
  return (unsigned int)f2bf(lo) | ((unsigned int)f2bf(hi) << 16);
}

// ---------------------------------------------------------------------------
// GEMM:  C = A[MxK] * W[KxN] + bias, A/W fp32 in global, converted to bf16.
// MODE 0: store Cf (f32) and Cb (bf16)        (Q projection)
// MODE 1: store Cb (bf16) only                (K / V projections)
// MODE 2: Cf = Res + relu(C)                  (FFN layer)
// Block: 256 threads (8 waves), tile 128x128, K-step 32.
// Wave (2 x 4 grid) computes 64x32 -> 4x2 wmma fragments.
// MODE is a template parameter: keeps the unrolled epilogue branch-free.
// ---------------------------------------------------------------------------
template <int MODE>
__global__ __launch_bounds__(256) void mab_gemm_kernel(
    const float* __restrict__ A, const float* __restrict__ W,
    const float* __restrict__ bias, const float* __restrict__ Res,
    float* __restrict__ Cf, unsigned short* __restrict__ Cb,
    int M, int K, int N)
{
  __shared__ unsigned int   As[128][16];   // 128 rows x 32 bf16 (pair-packed)
  __shared__ unsigned short Bs[128][32];   // [n][k] transposed, 32 bf16 per n

  const int tid  = threadIdx.x;
  const int lane = tid & 31;
  const int wid  = tid >> 5;
  const int waveM = wid >> 2;              // 0..1
  const int waveN = wid & 3;               // 0..3
  const int m0 = blockIdx.y * 128;
  const int n0 = blockIdx.x * 128;
  const int half = lane >> 4;              // lane half (0/1)
  const int l16  = lane & 15;
  const int c4   = half * 4;               // A-frag uint offset (K=8..15 upper half)
  const int boff = half * 8;               // B-frag: K=8h..8h+7 and 16+8h..16+8h+7

  FFrag acc[4][2];
#pragma unroll
  for (int mi = 0; mi < 4; ++mi)
#pragma unroll
    for (int ni = 0; ni < 2; ++ni)
#pragma unroll
      for (int v = 0; v < 8; ++v) acc[mi][ni].f[v] = 0.f;

  const int arow = tid >> 1, afb = (tid & 1) * 16;   // A tile: 16 floats/thread
  const int wk   = tid >> 3, wnb = (tid & 7) * 16;   // W tile: 16 floats/thread

  for (int kt = 0; kt < K; kt += 32) {
    const float* ap = A + (size_t)(m0 + arow) * K + kt + afb;
    const float* wp = W + (size_t)(kt + wk) * N + n0 + wnb;
    if (kt + 32 < K) {                      // prefetch next K-step tiles
      __builtin_prefetch(ap + 32, 0, 3);    // -> global_prefetch_b8
      __builtin_prefetch(wp + (size_t)32 * N, 0, 3);
    }
#pragma unroll
    for (int j = 0; j < 4; ++j) {
      F4 f = *(const F4*)(ap + j * 4);
      As[arow][(afb >> 1) + j * 2 + 0] = pack_bf2(f.x, f.y);
      As[arow][(afb >> 1) + j * 2 + 1] = pack_bf2(f.z, f.w);
    }
#pragma unroll
    for (int j = 0; j < 4; ++j) {
      F4 f = *(const F4*)(wp + j * 4);
      Bs[wnb + j * 4 + 0][wk] = f2bf(f.x);
      Bs[wnb + j * 4 + 1][wk] = f2bf(f.y);
      Bs[wnb + j * 4 + 2][wk] = f2bf(f.z);
      Bs[wnb + j * 4 + 3][wk] = f2bf(f.w);
    }
    __syncthreads();

    BFrag af[4], bf[2];
#pragma unroll
    for (int mi = 0; mi < 4; ++mi) {
      const int r = waveM * 64 + mi * 16 + l16;
      af[mi].q[0] = *(const U4*)&As[r][c4];
      af[mi].q[1] = *(const U4*)&As[r][8 + c4];
    }
#pragma unroll
    for (int ni = 0; ni < 2; ++ni) {
      const int n = waveN * 32 + ni * 16 + l16;
      bf[ni].q[0] = *(const U4*)&Bs[n][boff];
      bf[ni].q[1] = *(const U4*)&Bs[n][boff + 16];
    }
#pragma unroll
    for (int mi = 0; mi < 4; ++mi)
#pragma unroll
      for (int ni = 0; ni < 2; ++ni)
        acc[mi][ni].v = __builtin_amdgcn_wmma_f32_16x16x32_bf16(
            false, af[mi].v, false, bf[ni].v, (short)0, acc[mi][ni].v,
            false, false);
    __syncthreads();
  }

#pragma unroll
  for (int ni = 0; ni < 2; ++ni) {
    const int col = n0 + waveN * 32 + ni * 16 + l16;
    const float bv = bias[col];
#pragma unroll
    for (int mi = 0; mi < 4; ++mi) {
      const int rbase = m0 + waveM * 64 + mi * 16 + half * 8;
      size_t idx = (size_t)rbase * N + col;
#pragma unroll
      for (int v = 0; v < 8; ++v, idx += (size_t)N) {
        const float val = acc[mi][ni].f[v] + bv;
        if (MODE == 0)      { Cf[idx] = val; Cb[idx] = f2bf(val); }
        else if (MODE == 1) { Cb[idx] = f2bf(val); }
        else                { Cf[idx] = Res[idx] + fmaxf(val, 0.f); }
      }
    }
  }
}

// ---------------------------------------------------------------------------
// Flash attention: per (128-query block, head). Each wave owns 16 query rows.
// S = Q Kt * 1/32 -> online softmax -> O = P V ; out = Qf + O / l.
// ---------------------------------------------------------------------------
__global__ __launch_bounds__(256) void mab_attn_kernel(
    const unsigned short* __restrict__ Qb, const unsigned short* __restrict__ Kb,
    const unsigned short* __restrict__ Vb, const float* __restrict__ Qf,
    float* __restrict__ O, int Nq, int Nk)
{
  __shared__ unsigned short Ks[64][64];      // [key][feature]
  __shared__ unsigned short Vs[64][64];      // [feature][key] (transposed)
  __shared__ unsigned short Ps[8][16][64];   // per-wave P staging (C -> A layout)

  const int D   = 1024;
  const int h   = blockIdx.y;
  const int q0  = blockIdx.x * 128;
  const int tid = threadIdx.x, lane = tid & 31, wid = tid >> 5;
  const int half = lane >> 4, l16 = lane & 15;
  const int koff = half * 8;                 // A-frag K sub-offset
  const int boff = half * 8;                 // B-frag K offset
  const float scale = 0.03125f;              // 1/sqrt(1024)

  // Q fragments in A layout: 2 K-steps across the 64 head features
  BFrag qf[2];
  {
    const unsigned short* qp = Qb + (size_t)(q0 + wid * 16 + l16) * D + h * 64;
#pragma unroll
    for (int ks = 0; ks < 2; ++ks) {
      qf[ks].q[0] = *(const U4*)(qp + ks * 32 + koff);
      qf[ks].q[1] = *(const U4*)(qp + ks * 32 + 16 + koff);
    }
  }

  FFrag oacc[4];
#pragma unroll
  for (int fi = 0; fi < 4; ++fi)
#pragma unroll
    for (int v = 0; v < 8; ++v) oacc[fi].f[v] = 0.f;
  float mrow[8], lrow[8];
#pragma unroll
  for (int v = 0; v < 8; ++v) { mrow[v] = -INFINITY; lrow[v] = 0.f; }

  for (int kt = 0; kt < Nk; kt += 64) {
    // cooperative tile loads: 64 keys x 64 features
    const int key = tid >> 2, fb = (tid & 3) * 16;
    const unsigned short* kp = Kb + (size_t)(kt + key) * D + h * 64 + fb;
    *(U4*)&Ks[key][fb]     = *(const U4*)(kp);
    *(U4*)&Ks[key][fb + 8] = *(const U4*)(kp + 8);
    const unsigned short* vp = Vb + (size_t)(kt + key) * D + h * 64 + fb;
#pragma unroll
    for (int j = 0; j < 16; ++j) Vs[fb + j][key] = vp[j];   // transpose scatter
    __syncthreads();

    // S = Q * K^T  (4 column fragments of 16 keys each)
    FFrag s[4];
#pragma unroll
    for (int ni = 0; ni < 4; ++ni) {
#pragma unroll
      for (int v = 0; v < 8; ++v) s[ni].f[v] = 0.f;
#pragma unroll
      for (int ks = 0; ks < 2; ++ks) {
        BFrag kf;
        const int kk = ni * 16 + l16;
        kf.q[0] = *(const U4*)&Ks[kk][ks * 32 + boff];
        kf.q[1] = *(const U4*)&Ks[kk][ks * 32 + boff + 16];
        s[ni].v = __builtin_amdgcn_wmma_f32_16x16x32_bf16(
            false, qf[ks].v, false, kf.v, (short)0, s[ni].v, false, false);
      }
    }
#pragma unroll
    for (int ni = 0; ni < 4; ++ni)
#pragma unroll
      for (int v = 0; v < 8; ++v) s[ni].f[v] *= scale;

    // online softmax: row = v + 8*half, cols striped over 16-lane group
#pragma unroll
    for (int v = 0; v < 8; ++v) {
      float m = fmaxf(fmaxf(s[0].f[v], s[1].f[v]), fmaxf(s[2].f[v], s[3].f[v]));
#pragma unroll
      for (int d = 1; d < 16; d <<= 1) m = fmaxf(m, __shfl_xor(m, d, 32));
      const float mnew  = fmaxf(mrow[v], m);
      const float alpha = __expf(mrow[v] - mnew);
      mrow[v] = mnew;
      float rs = 0.f;
#pragma unroll
      for (int ni = 0; ni < 4; ++ni) {
        const float p = __expf(s[ni].f[v] - mnew);
        s[ni].f[v] = p;
        rs += p;
      }
#pragma unroll
      for (int d = 1; d < 16; d <<= 1) rs += __shfl_xor(rs, d, 32);
      lrow[v] = lrow[v] * alpha + rs;
#pragma unroll
      for (int fi = 0; fi < 4; ++fi) oacc[fi].f[v] *= alpha;
    }

    // P (C layout) -> per-wave LDS -> A layout fragments
#pragma unroll
    for (int ni = 0; ni < 4; ++ni)
#pragma unroll
      for (int v = 0; v < 8; ++v)
        Ps[wid][v + half * 8][ni * 16 + l16] = f2bf(s[ni].f[v]);
    asm volatile("s_wait_dscnt 0x0" ::: "memory");   // wave-local LDS RAW

    BFrag pf[2];
#pragma unroll
    for (int kp2 = 0; kp2 < 2; ++kp2) {
      pf[kp2].q[0] = *(const U4*)&Ps[wid][l16][kp2 * 32 + koff];
      pf[kp2].q[1] = *(const U4*)&Ps[wid][l16][kp2 * 32 + 16 + koff];
    }
#pragma unroll
    for (int fi = 0; fi < 4; ++fi) {
#pragma unroll
      for (int kp2 = 0; kp2 < 2; ++kp2) {
        BFrag vf;
        const int ft = fi * 16 + l16;
        vf.q[0] = *(const U4*)&Vs[ft][kp2 * 32 + boff];
        vf.q[1] = *(const U4*)&Vs[ft][kp2 * 32 + boff + 16];
        oacc[fi].v = __builtin_amdgcn_wmma_f32_16x16x32_bf16(
            false, pf[kp2].v, false, vf.v, (short)0, oacc[fi].v, false, false);
      }
    }
    __syncthreads();
  }

  // normalize + projected-Q residual
#pragma unroll
  for (int fi = 0; fi < 4; ++fi) {
    const int col = h * 64 + fi * 16 + l16;
#pragma unroll
    for (int v = 0; v < 8; ++v) {
      const int row = q0 + wid * 16 + v + half * 8;
      const size_t idx = (size_t)row * D + col;
      O[idx] = Qf[idx] + oacc[fi].f[v] / lrow[v];
    }
  }
}

// ---------------------------------------------------------------------------
// Row LayerNorm, D = 1024, one 256-thread block per row (4 elems/thread).
// ---------------------------------------------------------------------------
__global__ __launch_bounds__(256) void mab_ln_kernel(
    const float* __restrict__ X, const float* __restrict__ g,
    const float* __restrict__ b, float* __restrict__ Y)
{
  __shared__ float s1s[8], s2s[8];
  const int D = 1024;
  const int row = blockIdx.x, tid = threadIdx.x;
  const float* x = X + (size_t)row * D;
  const F4 f = *(const F4*)(x + tid * 4);
  float s1 = f.x + f.y + f.z + f.w;
  float s2 = f.x * f.x + f.y * f.y + f.z * f.z + f.w * f.w;
#pragma unroll
  for (int d = 1; d < 32; d <<= 1) {
    s1 += __shfl_xor(s1, d, 32);
    s2 += __shfl_xor(s2, d, 32);
  }
  const int lane = tid & 31, wid = tid >> 5;
  if (lane == 0) { s1s[wid] = s1; s2s[wid] = s2; }
  __syncthreads();
  float a1 = 0.f, a2 = 0.f;
#pragma unroll
  for (int i = 0; i < 8; ++i) { a1 += s1s[i]; a2 += s2s[i]; }
  const float mean = a1 * (1.f / 1024.f);
  const float var  = a2 * (1.f / 1024.f) - mean * mean;
  const float inv  = rsqrtf(var + 1e-5f);
  const int c = tid * 4;
  const F4 go = *(const F4*)(g + c);
  const F4 bo = *(const F4*)(b + c);
  F4 o;
  o.x = (f.x - mean) * inv * go.x + bo.x;
  o.y = (f.y - mean) * inv * go.y + bo.y;
  o.z = (f.z - mean) * inv * go.z + bo.z;
  o.w = (f.w - mean) * inv * go.w + bo.w;
  *(F4*)(Y + (size_t)row * D + c) = o;
}

// ---------------------------------------------------------------------------
extern "C" void kernel_launch(void* const* d_in, const int* in_sizes, int n_in,
                              void* d_out, int out_size, void* d_ws, size_t ws_size,
                              hipStream_t stream)
{
  (void)in_sizes; (void)n_in; (void)out_size; (void)ws_size;
  const float* query = (const float*)d_in[0];
  const float* key_  = (const float*)d_in[1];
  const float* Wq = (const float*)d_in[2];
  const float* bq = (const float*)d_in[3];
  const float* Wk = (const float*)d_in[4];
  const float* bk = (const float*)d_in[5];
  const float* Wv = (const float*)d_in[6];
  const float* bv = (const float*)d_in[7];
  const float* g1 = (const float*)d_in[8];
  const float* b1 = (const float*)d_in[9];
  const float* Wl = (const float*)d_in[10];
  const float* bl = (const float*)d_in[11];
  const float* g2 = (const float*)d_in[12];
  const float* b2 = (const float*)d_in[13];
  float* out = (float*)d_out;

  const int Nq = 4096, Nk = 4096, D = 1024, H = 16;

  char* w = (char*)d_ws;
  float* Qf   = (float*)w;                 w += (size_t)Nq * D * sizeof(float);
  float* Obuf = (float*)w;                 w += (size_t)Nq * D * sizeof(float);
  float* ln1  = (float*)w;                 w += (size_t)Nq * D * sizeof(float);
  float* Ybuf = (float*)w;                 w += (size_t)Nq * D * sizeof(float);
  unsigned short* Qb = (unsigned short*)w; w += (size_t)Nq * D * sizeof(unsigned short);
  unsigned short* Kb = (unsigned short*)w; w += (size_t)Nk * D * sizeof(unsigned short);
  unsigned short* Vb = (unsigned short*)w; w += (size_t)Nk * D * sizeof(unsigned short);

  const dim3 blk(256);
  const dim3 gGemm(D / 128, Nq / 128);

  // projections
  mab_gemm_kernel<0><<<gGemm, blk, 0, stream>>>(query, Wq, bq, nullptr, Qf, Qb,
                                                Nq, D, D);
  mab_gemm_kernel<1><<<gGemm, blk, 0, stream>>>(key_, Wk, bk, nullptr, nullptr, Kb,
                                                Nk, D, D);
  mab_gemm_kernel<1><<<gGemm, blk, 0, stream>>>(key_, Wv, bv, nullptr, nullptr, Vb,
                                                Nk, D, D);
  // attention + residual
  const dim3 gAttn(Nq / 128, H);
  mab_attn_kernel<<<gAttn, blk, 0, stream>>>(Qb, Kb, Vb, Qf, Obuf, Nq, Nk);
  // LN1
  mab_ln_kernel<<<dim3(Nq), blk, 0, stream>>>(Obuf, g1, b1, ln1);
  // FFN: y = ln1 + relu(ln1 @ Wl + bl)
  mab_gemm_kernel<2><<<gGemm, blk, 0, stream>>>(ln1, Wl, bl, ln1, Ybuf, nullptr,
                                                Nq, D, D);
  // LN2 -> output
  mab_ln_kernel<<<dim3(Nq), blk, 0, stream>>>(Ybuf, g2, b2, out);
}
